// MoELoRAQuantLinear_39943195853510
// MI455X (gfx1250) — compile-verified, matrix operand order checked by reading, not run on hardware
//
#include <hip/hip_runtime.h>

typedef _Float16 v16h __attribute__((ext_vector_type(16)));
typedef _Float16 v8h  __attribute__((ext_vector_type(8)));
typedef float    v8f  __attribute__((ext_vector_type(8)));
typedef int      v4i  __attribute__((vector_size(4 * sizeof(int))));  // gcc-style, matches builtin

#define NTOK    8192
#define DIN     2048
#define DOUT    2048
#define NEXP    8
#define LRANK   16
#define JDIM    128          // NEXP * LRANK
#define KTOT    (DIN + JDIM) // extended K: base GEMM + folded LoRA-B GEMM
#define NK      (KTOT / 32)
#define SCALING 2.0f
#define TS      40           // padded LDS row stride in f16 (80 B, 16B-aligned)

#define AS1 __attribute__((address_space(1)))
#define AS3 __attribute__((address_space(3)))

// ---------------------------------------------------------------------------
// CDNA5 async global->LDS path (ASYNCcnt), with sync fallback.
// Builtin signature (probe-confirmed by round-2 diagnostic):
//   void __builtin_amdgcn_global_load_async_to_lds_b128(
//       int4 addrspace(1)* gsrc, int4 addrspace(3)* ldst, imm offset, imm cpol)
// ---------------------------------------------------------------------------
#if __has_builtin(__builtin_amdgcn_global_load_async_to_lds_b128) && \
    __has_builtin(__builtin_amdgcn_s_wait_asynccnt)
#define HAVE_ASYNC_LDS 1
#define ASYNC_CP16F16(ldst, gsrc)                         \
    __builtin_amdgcn_global_load_async_to_lds_b128(       \
        (AS1 v4i*)(gsrc), (AS3 v4i*)(ldst), 0, 0)
#define WAIT_ASYNC(n) __builtin_amdgcn_s_wait_asynccnt(n)
#else
#define HAVE_ASYNC_LDS 0
#define ASYNC_CP16F16(ldst, gsrc) \
    do { *(v8h*)(ldst) = *(const v8h*)(gsrc); } while (0)
#define WAIT_ASYNC(n) do {} while (0)
#endif

// Stage 16 f16 (32 B) for A and B tiles: 4 async b128 per thread per tile,
// i.e. exactly 4 ASYNCcnt increments per wave per tile.
__device__ __forceinline__ void stage_tile(_Float16* lA, const _Float16* gA,
                                           _Float16* lB, const _Float16* gB) {
    ASYNC_CP16F16(lA, gA);
    ASYNC_CP16F16(lA + 8, gA + 8);
    ASYNC_CP16F16(lB, gB);
    ASYNC_CP16F16(lB + 8, gB + 8);
}

// ---------------------------------------------------------------------------
// helpers
// ---------------------------------------------------------------------------

// Convert 16 consecutive f32 -> 16 f16 into LDS (two b128 stores).
__device__ __forceinline__ void stage16_f32(_Float16* dst, const float* src) {
    const float4* s = (const float4*)src;
    float4 f0 = s[0], f1 = s[1], f2 = s[2], f3 = s[3];
    v8h a, b;
    a[0] = (_Float16)f0.x; a[1] = (_Float16)f0.y; a[2] = (_Float16)f0.z; a[3] = (_Float16)f0.w;
    a[4] = (_Float16)f1.x; a[5] = (_Float16)f1.y; a[6] = (_Float16)f1.z; a[7] = (_Float16)f1.w;
    b[0] = (_Float16)f2.x; b[1] = (_Float16)f2.y; b[2] = (_Float16)f2.z; b[3] = (_Float16)f2.w;
    b[4] = (_Float16)f3.x; b[5] = (_Float16)f3.y; b[6] = (_Float16)f3.z; b[7] = (_Float16)f3.w;
    *(v8h*)dst       = a;
    *(v8h*)(dst + 8) = b;
}

// Load one 16x32 f16 WMMA A/B fragment row for this lane from a K-contiguous
// LDS row. Per ISA 16-bit layout: elems 0..7 <-> K = 8*half + i,
// elems 8..15 <-> K = 16 + 8*half + i.  Two ds_load_b128.
__device__ __forceinline__ v16h load_frag(const _Float16* rowp, int hf) {
    v8h lo = *(const v8h*)(rowp + 8 * hf);
    v8h hi = *(const v8h*)(rowp + 16 + 8 * hf);
    v16h r;
#pragma unroll
    for (int i = 0; i < 8; ++i) { r[i] = lo[i]; r[i + 8] = hi[i]; }
    return r;
}

#define WMMA_F16(A, B, C) \
    __builtin_amdgcn_wmma_f32_16x16x32_f16(false, (A), false, (B), (short)0, (C), false, false)

// ---------------------------------------------------------------------------
// Kernel 0a: elementwise f32 -> f16 (8 elems / thread)
// ---------------------------------------------------------------------------
__global__ __launch_bounds__(256) void cvt_f16_kernel(
    const float* __restrict__ src, _Float16* __restrict__ dst, int n) {
    int i = (blockIdx.x * 256 + threadIdx.x) * 8;
    if (i >= n) return;
    const float4* s = (const float4*)(src + i);
    float4 a = s[0], b = s[1];
    v8h o;
    o[0] = (_Float16)a.x; o[1] = (_Float16)a.y; o[2] = (_Float16)a.z; o[3] = (_Float16)a.w;
    o[4] = (_Float16)b.x; o[5] = (_Float16)b.y; o[6] = (_Float16)b.z; o[7] = (_Float16)b.w;
    *(v8h*)(dst + i) = o;
}

// ---------------------------------------------------------------------------
// Kernel 0b: lora_B [JDIM][DOUT] f32 -> lBt [DOUT][JDIM] f16 (transpose)
// One thread per output column o; coalesced reads across o.
// ---------------------------------------------------------------------------
__global__ __launch_bounds__(256) void transpose_lB_kernel(
    const float* __restrict__ lB, _Float16* __restrict__ lBt) {
    int o = blockIdx.x * 256 + threadIdx.x;
    if (o >= DOUT) return;
    _Float16* dst = lBt + (size_t)o * JDIM;
#pragma unroll 4
    for (int j = 0; j < JDIM; ++j) dst[j] = (_Float16)lB[(size_t)j * DOUT + o];
}

// ---------------------------------------------------------------------------
// Kernel 1: router -> dense normalized top-2 gate [NTOK, NEXP] (f32, in ws)
// One wave32 per token.
// ---------------------------------------------------------------------------
__global__ __launch_bounds__(256) void moe_router_kernel(
    const float* __restrict__ x, const float* __restrict__ rw,
    const float* __restrict__ rb, float* __restrict__ gate) {
    int t    = threadIdx.x;
    int lane = t & 31;
    int n    = (blockIdx.x * 256 + t) >> 5;
    if (n >= NTOK) return;

    const float* xr = x + (size_t)n * DIN;
    float p[NEXP];
#pragma unroll
    for (int e = 0; e < NEXP; ++e) p[e] = 0.f;
    for (int d = lane; d < DIN; d += 32) {
        float xv = xr[d];
#pragma unroll
        for (int e = 0; e < NEXP; ++e) p[e] += xv * rw[e * DIN + d];
    }
#pragma unroll
    for (int e = 0; e < NEXP; ++e) {
#pragma unroll
        for (int off = 16; off > 0; off >>= 1) p[e] += __shfl_xor(p[e], off, 32);
    }
    if (lane == 0) {
        float lg[NEXP];
#pragma unroll
        for (int e = 0; e < NEXP; ++e) lg[e] = p[e] + rb[e];
        int i1 = 0;
#pragma unroll
        for (int e = 1; e < NEXP; ++e) if (lg[e] > lg[i1]) i1 = e;
        int i2 = (i1 == 0) ? 1 : 0;
#pragma unroll
        for (int e = 0; e < NEXP; ++e) if (e != i1 && lg[e] > lg[i2]) i2 = e;
        float e2 = __expf(lg[i2] - lg[i1]);
        float w1 = 1.f / (1.f + e2);
        float w2 = e2 * w1;
        float* gp = gate + (size_t)n * NEXP;
#pragma unroll
        for (int e = 0; e < NEXP; ++e) gp[e] = 0.f;
        gp[i1] = w1;
        gp[i2] = w2;
    }
}

// ---------------------------------------------------------------------------
// Kernel 2: mid_w[n, e*16+r] = f16( SCALING * gate[n,e] * (x @ lora_A)[n,e,r] )
// WMMA GEMM [NTOK x DIN] x [DIN x JDIM]. 128-token row blocks.
// ---------------------------------------------------------------------------
__global__ __launch_bounds__(256) void moe_mid_kernel(
    const float* __restrict__ x, const float* __restrict__ loraA,
    const float* __restrict__ gate, _Float16* __restrict__ midw) {
    __shared__ __align__(16) _Float16 sA[128 * TS];
    __shared__ __align__(16) _Float16 sB[JDIM * TS];

    int t    = threadIdx.x;
    int n0   = blockIdx.x * 128;
    int row  = t >> 1;
    int half = t & 1;
    int kk0  = half * 16;

    int wave = t >> 5, lane = t & 31;
    int hf = lane >> 4, ln = lane & 15;
    int wm = (wave >> 2) * 64;
    int wn = (wave & 3) * 32;

    v8f acc[4][2];
#pragma unroll
    for (int i = 0; i < 4; ++i)
#pragma unroll
        for (int j = 0; j < 2; ++j) acc[i][j] = (v8f){};

#pragma unroll 1
    for (int k0 = 0; k0 < DIN; k0 += 32) {
        stage16_f32(&sA[row * TS + kk0], x + (size_t)(n0 + row) * DIN + k0 + kk0);
        {
            int e = row >> 4, r = row & 15;
            const float* src = loraA + ((size_t)e * DIN + (k0 + kk0)) * LRANK + r;
            _Float16* d = &sB[row * TS + kk0];
#pragma unroll
            for (int i = 0; i < 16; ++i) d[i] = (_Float16)src[(size_t)i * LRANK];
        }
        __syncthreads();

        v16h af[4], bf[2];
#pragma unroll
        for (int i = 0; i < 4; ++i) af[i] = load_frag(&sA[(wm + 16 * i + ln) * TS], hf);
#pragma unroll
        for (int j = 0; j < 2; ++j) bf[j] = load_frag(&sB[(wn + 16 * j + ln) * TS], hf);
#pragma unroll
        for (int i = 0; i < 4; ++i)
#pragma unroll
            for (int j = 0; j < 2; ++j) acc[i][j] = WMMA_F16(af[i], bf[j], acc[i][j]);
        __syncthreads();
    }

#pragma unroll
    for (int i = 0; i < 4; ++i) {
#pragma unroll
        for (int j = 0; j < 2; ++j) {
            int jc = wn + 16 * j + ln;
            int e  = jc >> 4;
#pragma unroll
            for (int v = 0; v < 8; ++v) {
                int n = n0 + wm + 16 * i + v + 8 * hf;
                midw[(size_t)n * JDIM + jc] =
                    (_Float16)(SCALING * gate[(size_t)n * NEXP + e] * acc[i][j][v]);
            }
        }
    }
}

// ---------------------------------------------------------------------------
// Kernel 3: out = xh @ bwh^T + base_b + midw @ lBt^T  (all operands f16)
// Extended-K WMMA GEMM, double-buffered LDS, async global->LDS staging.
// Grid: (NTOK/128) x (DOUT/128) blocks flattened; 8 waves / block.
// ---------------------------------------------------------------------------
__global__ __launch_bounds__(256) void moe_main_kernel(
    const _Float16* __restrict__ xh, const _Float16* __restrict__ bwh,
    const float* __restrict__ bb, const _Float16* __restrict__ lBt,
    const _Float16* __restrict__ midw, float* __restrict__ out) {
    __shared__ __align__(16) _Float16 sA[2][128 * TS];
    __shared__ __align__(16) _Float16 sB[2][128 * TS];

    int t  = threadIdx.x;
    int bm = blockIdx.x >> 4;          // DOUT/128 == 16 column blocks
    int bn = blockIdx.x & 15;
    int n0 = bm * 128;                 // token base
    int c0 = bn * 128;                 // output-column base

    int row  = t >> 1;
    int half = t & 1;
    int kk0  = half * 16;

    int wave = t >> 5, lane = t & 31;
    int hf = lane >> 4, ln = lane & 15;
    int wm = (wave >> 2) * 64;
    int wn = (wave & 3) * 32;

    // K-phase source selectors (base phase: xh/bwh; LoRA tail: midw/lBt)
    auto srcA = [&](int k0) -> const _Float16* {
        return (k0 < DIN) ? xh + (size_t)(n0 + row) * DIN + k0 + kk0
                          : midw + (size_t)(n0 + row) * JDIM + (k0 - DIN) + kk0;
    };
    auto srcB = [&](int k0) -> const _Float16* {
        return (k0 < DIN) ? bwh + (size_t)(c0 + row) * DIN + k0 + kk0
                          : lBt + (size_t)(c0 + row) * JDIM + (k0 - DIN) + kk0;
    };

    v8f acc[4][2];
#pragma unroll
    for (int i = 0; i < 4; ++i)
#pragma unroll
        for (int j = 0; j < 2; ++j) acc[i][j] = (v8f){};

    // prologue: async-stage tile 0 into buffer 0 (4 ASYNCcnt per wave)
    stage_tile(&sA[0][row * TS + kk0], srcA(0), &sB[0][row * TS + kk0], srcB(0));

#pragma unroll 1
    for (int k = 0; k < NK; ++k) {
        int b = k & 1;
        if (k + 1 < NK) {
            // prefetch next tile into the other buffer while tile k completes
            stage_tile(&sA[1 - b][row * TS + kk0], srcA((k + 1) * 32),
                       &sB[1 - b][row * TS + kk0], srcB((k + 1) * 32));
            WAIT_ASYNC(4);   // in-order: oldest 4 (tile k) done; tile k+1 in flight
        } else {
            WAIT_ASYNC(0);
        }
        __syncthreads();     // tile k visible from all waves

        v16h af[4], bf[2];
#pragma unroll
        for (int i = 0; i < 4; ++i) af[i] = load_frag(&sA[b][(wm + 16 * i + ln) * TS], hf);
#pragma unroll
        for (int j = 0; j < 2; ++j) bf[j] = load_frag(&sB[b][(wn + 16 * j + ln) * TS], hf);
#pragma unroll
        for (int i = 0; i < 4; ++i)
#pragma unroll
            for (int j = 0; j < 2; ++j) acc[i][j] = WMMA_F16(af[i], bf[j], acc[i][j]);
        __syncthreads();     // everyone done reading buffer b before it is refilled
    }

#pragma unroll
    for (int i = 0; i < 4; ++i) {
#pragma unroll
        for (int j = 0; j < 2; ++j) {
            int gn = c0 + wn + 16 * j + ln;
            float bias = bb[gn];
#pragma unroll
            for (int v = 0; v < 8; ++v) {
                int gm = n0 + wm + 16 * i + v + 8 * hf;
                out[(size_t)gm * DOUT + gn] = acc[i][j][v] + bias;
            }
        }
    }
}

// ---------------------------------------------------------------------------
extern "C" void kernel_launch(void* const* d_in, const int* in_sizes, int n_in,
                              void* d_out, int out_size, void* d_ws, size_t ws_size,
                              hipStream_t stream) {
    (void)in_sizes; (void)n_in; (void)out_size; (void)ws_size;
    const float* x   = (const float*)d_in[0];
    const float* bw  = (const float*)d_in[1];
    const float* bb  = (const float*)d_in[2];
    const float* rw  = (const float*)d_in[3];
    const float* rb  = (const float*)d_in[4];
    const float* lA  = (const float*)d_in[5];
    const float* lB  = (const float*)d_in[6];
    float*       out = (float*)d_out;

    // workspace layout (~43 MB):
    char* ws = (char*)d_ws;
    float*    gate = (float*)ws;     ws += (size_t)NTOK * NEXP * sizeof(float);     // 256 KB
    _Float16* midw = (_Float16*)ws;  ws += (size_t)NTOK * JDIM * sizeof(_Float16);  //   2 MB
    _Float16* xh   = (_Float16*)ws;  ws += (size_t)NTOK * DIN * sizeof(_Float16);   //  32 MB
    _Float16* bwh  = (_Float16*)ws;  ws += (size_t)DOUT * DIN * sizeof(_Float16);   //   8 MB
    _Float16* lBt  = (_Float16*)ws;                                                 // 512 KB

    cvt_f16_kernel<<<(NTOK * DIN) / (8 * 256), 256, 0, stream>>>(x, xh, NTOK * DIN);
    cvt_f16_kernel<<<(DOUT * DIN) / (8 * 256), 256, 0, stream>>>(bw, bwh, DOUT * DIN);
    transpose_lB_kernel<<<DOUT / 256, 256, 0, stream>>>(lB, lBt);
    moe_router_kernel<<<NTOK / 8, 256, 0, stream>>>(x, rw, rb, gate);
    moe_mid_kernel<<<NTOK / 128, 256, 0, stream>>>(x, lA, gate, midw);
    moe_main_kernel<<<(NTOK / 128) * (DOUT / 128), 256, 0, stream>>>(xh, bwh, bb, lBt, midw, out);
}
